// PAdicMemory_23295902613804
// MI455X (gfx1250) — matrix operand order repaired and checked by reading, not compiled, not run on hardware
//
#include <hip/hip_runtime.h>
#include <hip/hip_bf16.h>
#include <hip/hip_fp16.h>

// ---- CDNA5 WMMA fused tree-router kernel (MI455X / gfx1250, wave32) ----

typedef __attribute__((ext_vector_type(16))) _Float16 v16h;
typedef __attribute__((ext_vector_type(8)))  _Float16 v8h;
typedef __attribute__((ext_vector_type(4)))  _Float16 v4h;
typedef __attribute__((ext_vector_type(8)))  float    v8f;
typedef __attribute__((ext_vector_type(4)))  float    v4f;

#define D_MODEL   1024
#define M_LEAF    256
#define N_INT     255
#define BT_TOKENS 32768          // 8 * 4096
#define ROWS      64             // tokens per block (4 row-tiles of 16)
#define RT_TILES  4
#define NWAVES    8
#define XS_STRIDE 1032           // 1024 + 8 halves pad (bank-conflict avoidance)
#define QS_STRIDE 1032
#define GS_STRIDE 264            // floats
#define LP_STRIDE 264            // halves

#define WMMA_F16(a, b, c) \
  __builtin_amdgcn_wmma_f32_16x16x32_f16(false, (a), false, (b), (short)0, (c), false, false)

// ---------------- weight conversion / layout kernel ----------------
__global__ void convert_weights(const float* __restrict__ Wq,
                                const float* __restrict__ routers,
                                const float* __restrict__ lv,
                                const float* __restrict__ wout,
                                _Float16* __restrict__ Wq_h,
                                _Float16* __restrict__ r_h,
                                _Float16* __restrict__ lvT,
                                _Float16* __restrict__ wo_h) {
  int i = blockIdx.x * 256 + threadIdx.x;
  if (i < D_MODEL * D_MODEL) {
    Wq_h[i] = (_Float16)Wq[i];
    wo_h[i] = (_Float16)wout[i];
  }
  if (i < 256 * D_MODEL) {
    int r = i >> 10;
    r_h[i] = (r < N_INT) ? (_Float16)routers[i] : (_Float16)0.0f;
  }
  if (i < M_LEAF * D_MODEL) {
    int m = i >> 10;
    int d = i & 1023;
    lvT[d * M_LEAF + m] = (_Float16)lv[i];
  }
}

// ---------------- fused per-token-tile kernel ----------------
__global__ __launch_bounds__(256, 1) void fused_tree_router(
    const float*    __restrict__ x,      // (BT, 1024) f32
    const _Float16* __restrict__ Wq_h,   // (1024, 1024) (e,d)
    const _Float16* __restrict__ r_h,    // (256, 1024)  (n,d), row255 = 0
    const _Float16* __restrict__ lvT,    // (1024, 256)  (d,m)
    const _Float16* __restrict__ wo_h,   // (1024, 1024) (e,d)
    const float*    __restrict__ bias,   // (1024,) f32
    float*          __restrict__ out) {  // (BT, 1024) f32
  // Region 0 (132 KB): Xs during phase 1, then aliased by Gs (f32) + LPs.
  __shared__ _Float16 Xs[ROWS * XS_STRIDE];   // 132,096 B
  // Region 1 (132 KB): Qs during phases 1-2, then ret during phases 3-4.
  __shared__ _Float16 Qs[ROWS * QS_STRIDE];   // 132,096 B

  float*    Gs  = reinterpret_cast<float*>(Xs);       // 64*264 f32 = 67,584 B
  _Float16* LPs = Xs + (ROWS * GS_STRIDE * 2);        // 64*264 f16 = 33,792 B

  const int tid   = threadIdx.x;
  const int lane  = tid & 31;
  const int wid   = tid >> 5;
  const int lmod  = lane & 15;     // N (or M) index within tile
  const int lhalf = lane >> 4;     // 0 or 1 (lane group)

  const int r0 = blockIdx.x * ROWS;  // global token base row

  // A-matrix fragment (16x32 f16) from LDS.
  auto loadA = [&](const _Float16* base, int stride, int row0, int k0) -> v16h {
    const _Float16* p = base + (row0 + lmod) * stride + k0 + lhalf * 16;
    v8h lo = *(const v8h*)(p);
    v8h hi = *(const v8h*)(p + 8);
    v16h a;
#pragma unroll
    for (int i = 0; i < 8; ++i) { a[i] = lo[i]; a[i + 8] = hi[i]; }
    return a;
  };
  // B-matrix fragment (32x16 f16) from global (N-major weight matrix).
  auto loadB = [&](const _Float16* __restrict__ base, int stride, int col0,
                   int k0) -> v16h {
    return *(const v16h*)(base + (col0 + lmod) * stride + k0 + lhalf * 16);
  };
  // Store one 16x16 f32 accumulator as f16 into an LDS matrix.
  auto storeC = [&](v8f acc, _Float16* base, int stride, int row0, int col0) {
#pragma unroll
    for (int r = 0; r < 8; ++r) {
      int row = row0 + r + lhalf * 8;
      base[row * stride + col0 + lmod] = (_Float16)acc[r];
    }
  };

  // ---------------- Phase 0: stage X tile as f16 in LDS ----------------
  for (int i = tid; i < ROWS * (D_MODEL / 4); i += 256) {
    int row = i >> 8;          // 0..63
    int c4  = i & 255;         // float4 column
    v4f v = *(const v4f*)(x + (size_t)(r0 + row) * D_MODEL + c4 * 4);
    v4h h;
#pragma unroll
    for (int j = 0; j < 4; ++j) h[j] = (_Float16)v[j];
    *(v4h*)(&Xs[row * XS_STRIDE + c4 * 4]) = h;
  }
  __syncthreads();

  // --- Phase 1: Q = X * Wq^T (K = 1024), e-tile pair x 4 row-tiles ---
  // Each k-step: 4 A fragments (LDS) + 2 B fragments (L2) feed 8 independent
  // WMMA chains -> 0.5 KB A-traffic and 0.25 KB B-traffic per WMMA.
  for (int et = wid; et < 32; et += NWAVES) {
    v8f acc0[RT_TILES] = {};
    v8f acc1[RT_TILES] = {};
#pragma unroll 2
    for (int k0 = 0; k0 < D_MODEL; k0 += 32) {
      v16h b0 = loadB(Wq_h, D_MODEL, et * 16, k0);
      v16h b1 = loadB(Wq_h, D_MODEL, (et + 32) * 16, k0);
#pragma unroll
      for (int rt = 0; rt < RT_TILES; ++rt) {
        v16h a = loadA(Xs, XS_STRIDE, rt * 16, k0);
        acc0[rt] = WMMA_F16(a, b0, acc0[rt]);
        acc1[rt] = WMMA_F16(a, b1, acc1[rt]);
      }
    }
#pragma unroll
    for (int rt = 0; rt < RT_TILES; ++rt) {
      storeC(acc0[rt], Qs, QS_STRIDE, rt * 16, et * 16);
      storeC(acc1[rt], Qs, QS_STRIDE, rt * 16, (et + 32) * 16);
    }
  }
  __syncthreads();

  // ---------------- Phase 2: gate = sigmoid(Q * routers^T) ----------------
  for (int et = wid; et < 16; et += NWAVES) {
    v8f acc[RT_TILES] = {};
#pragma unroll 2
    for (int k0 = 0; k0 < D_MODEL; k0 += 32) {
      v16h b = loadB(r_h, D_MODEL, et * 16, k0);
#pragma unroll
      for (int rt = 0; rt < RT_TILES; ++rt) {
        v16h a = loadA(Qs, QS_STRIDE, rt * 16, k0);
        acc[rt] = WMMA_F16(a, b, acc[rt]);
      }
    }
#pragma unroll
    for (int rt = 0; rt < RT_TILES; ++rt) {
#pragma unroll
      for (int r = 0; r < 8; ++r) {
        int row = rt * 16 + r + lhalf * 8;
        Gs[row * GS_STRIDE + et * 16 + lmod] =
            1.0f / (1.0f + __expf(-acc[rt][r]));
      }
    }
  }
  __syncthreads();

  // ------- Phase 2b: leaf probs = product of branch probs up the heap -------
  for (int i = tid; i < ROWS * M_LEAF; i += 256) {
    int tr = i >> 8;
    int m  = i & 255;
    unsigned id = 256u + (unsigned)m;      // 1-indexed heap leaf id
    const float* g = Gs + tr * GS_STRIDE;
    float p = 1.0f;
#pragma unroll
    for (int l = 0; l < 8; ++l) {
      unsigned bit = id & 1u;              // right child of parent?
      id >>= 1;                            // ancestor (internal node, 1-indexed)
      float gv = g[id - 1u];
      p *= bit ? gv : (1.0f - gv);
    }
    LPs[tr * LP_STRIDE + m] = (_Float16)p;
  }
  __syncthreads();

  // ------------- Phase 3: ret = lp * leaf_values  (K = 256) -------------
  for (int dt = wid; dt < 64; dt += NWAVES) {
    v8f acc[RT_TILES] = {};
#pragma unroll 2
    for (int k0 = 0; k0 < M_LEAF; k0 += 32) {
      v16h b = loadB(lvT, M_LEAF, dt * 16, k0);
#pragma unroll
      for (int rt = 0; rt < RT_TILES; ++rt) {
        v16h a = loadA(LPs, LP_STRIDE, rt * 16, k0);
        acc[rt] = WMMA_F16(a, b, acc[rt]);
      }
    }
#pragma unroll
    for (int rt = 0; rt < RT_TILES; ++rt)
      storeC(acc[rt], Qs, QS_STRIDE, rt * 16, dt * 16);   // ret reuses Qs
  }
  __syncthreads();

  // --- Phase 4: out = ret * Wout^T + b, e-tile pair x 4 row-tiles ---
  for (int et = wid; et < 32; et += NWAVES) {
    v8f acc0[RT_TILES] = {};
    v8f acc1[RT_TILES] = {};
#pragma unroll 2
    for (int k0 = 0; k0 < D_MODEL; k0 += 32) {
      v16h b0 = loadB(wo_h, D_MODEL, et * 16, k0);
      v16h b1 = loadB(wo_h, D_MODEL, (et + 32) * 16, k0);
#pragma unroll
      for (int rt = 0; rt < RT_TILES; ++rt) {
        v16h a = loadA(Qs, QS_STRIDE, rt * 16, k0);
        acc0[rt] = WMMA_F16(a, b0, acc0[rt]);
        acc1[rt] = WMMA_F16(a, b1, acc1[rt]);
      }
    }
    float bv0 = bias[et * 16 + lmod];
    float bv1 = bias[(et + 32) * 16 + lmod];
#pragma unroll
    for (int rt = 0; rt < RT_TILES; ++rt) {
#pragma unroll
      for (int r = 0; r < 8; ++r) {
        int row = r0 + rt * 16 + r + lhalf * 8;
        out[(size_t)row * D_MODEL + et * 16 + lmod] = acc0[rt][r] + bv0;
        out[(size_t)row * D_MODEL + (et + 32) * 16 + lmod] = acc1[rt][r] + bv1;
      }
    }
  }
}

// ---------------- host launcher ----------------
extern "C" void kernel_launch(void* const* d_in, const int* in_sizes, int n_in,
                              void* d_out, int out_size, void* d_ws,
                              size_t ws_size, hipStream_t stream) {
  const float* x       = (const float*)d_in[0];  // (8,4096,1024)
  const float* Wq      = (const float*)d_in[1];  // (1024,1024)
  const float* routers = (const float*)d_in[2];  // (255,1024)
  // d_in[3] = leaf_keys: unused by the reference
  const float* lv      = (const float*)d_in[4];  // (256,1024)
  const float* wout    = (const float*)d_in[5];  // (1024,1024)
  const float* wb      = (const float*)d_in[6];  // (1024,)
  float* out = (float*)d_out;

  _Float16* ws   = (_Float16*)d_ws;
  _Float16* Wq_h = ws;                     // 1,048,576 halves
  _Float16* wo_h = Wq_h + 1048576;         // 1,048,576 halves
  _Float16* r_h  = wo_h + 1048576;         //   262,144 halves (256 rows)
  _Float16* lvT  = r_h  + 262144;          //   262,144 halves
  // total ws usage: 5.25 MB

  convert_weights<<<(D_MODEL * D_MODEL + 255) / 256, 256, 0, stream>>>(
      Wq, routers, lv, wout, Wq_h, r_h, lvT, wo_h);

  fused_tree_router<<<BT_TOKENS / ROWS, 256, 0, stream>>>(
      x, Wq_h, r_h, lvT, wo_h, wb, out);
}
